// Dale_CBcell_54503134986287
// MI455X (gfx1250) — matrix operand order repaired and checked by reading, not compiled, not run on hardware
//
#include <hip/hip_runtime.h>
#include <math.h>

// ---------------------------------------------------------------------------
// Dale CB-cell update, CDNA5 (gfx1250) WMMA version.
//
//  prep kernel (1 block): builds two f16 "slab" matrices in workspace,
//    Az[48][64] = [ softplus(K) | P_z      | 0 ]
//    Av[48][64] = [ Dale(W)     | P*mask   | 0 ]
//  main kernel: per 16-batch tile,
//    B-frag = [ sigmoid(hidden^T) rows 0..47 ; x rows 0..7 ; 0 pad ]  (f16)
//    pre_z = Az @ Bfrag ; pre_v = Av @ Bfrag  via v_wmma_f32_16x16x32_f16
//    out   = (1 - DT*sigmoid(pre_z + b_z)) * v + DT*(pre_v + b_v)
//
//  Sigmoid uses v_exp_f32 + v_rcp_f32 (trans pipe, co-executes with WMMA)
//  instead of the IEEE divide sequence the '/' operator lowers to.
// ---------------------------------------------------------------------------

typedef __attribute__((ext_vector_type(16))) _Float16 v16h;
typedef __attribute__((ext_vector_type(8)))  _Float16 v8h;
typedef __attribute__((ext_vector_type(8)))  float    v8f;
typedef __attribute__((ext_vector_type(4)))  float    f32x4;

#define Hdim  48
#define INdim 8
#define KPAD  64
#define DTc   0.1f

__device__ __forceinline__ float sigf(float v) {
  // 1/(1+e^-v) via v_exp_f32 + v_rcp_f32 (no div_fixup/div_fmas sequence)
  return __builtin_amdgcn_rcpf(1.0f + __expf(-v));
}
__device__ __forceinline__ float softplusf(float v) {
  return (v > 15.0f) ? v : log1pf(__expf(v));
}

// ---------------------------------------------------------------------------
// Prep: fold softplus / Dale constraints / input mask into f16 slabs.
// ---------------------------------------------------------------------------
__global__ void dale_prep(const float* __restrict__ Kw,
                          const float* __restrict__ Cw,
                          const float* __restrict__ P,
                          const float* __restrict__ Pz,
                          const float* __restrict__ e_e,
                          const float* __restrict__ e_i,
                          _Float16* __restrict__ Az,
                          _Float16* __restrict__ Av) {
  const float ee = e_e[0];
  const float ei = e_i[0];
  for (int idx = threadIdx.x; idx < Hdim * KPAD; idx += blockDim.x) {
    const int h = idx / KPAD;
    const int k = idx - h * KPAD;
    float az = 0.0f, av = 0.0f;
    if (k < Hdim) {
      const float ksp = softplusf(Kw[h * Hdim + k]);
      const float csp = softplusf(Cw[h * Hdim + k]);
      const float S   = ksp + csp;
      az = ksp;
      // Dale: excitatory columns [0,24) clamped >= 0, inhibitory [24,48) <= 0
      av = (k < Hdim / 2) ? fmaxf(ee * S, 0.0f) : -fmaxf(-(ei * S), 0.0f);
    } else if (k < Hdim + INdim) {
      const int i = k - Hdim;
      az = Pz[h * INdim + i];
      const bool masked = ((h >= Hdim / 4) && (h < Hdim / 2)) || (h >= 3 * Hdim / 4);
      av = masked ? 0.0f : P[h * INdim + i];
    }
    Az[idx] = (_Float16)az;
    Av[idx] = (_Float16)av;
  }
}

// ---------------------------------------------------------------------------
// Main: one wave per 16-batch tile (grid-stride). 12 WMMAs per tile.
// ---------------------------------------------------------------------------
__global__ void __launch_bounds__(256) dale_main(
    const float* __restrict__ hidden,   // (B, 48) row-major
    const float* __restrict__ x,        // (8, B)  row-major
    const float* __restrict__ b_v,      // (48)
    const float* __restrict__ b_z,      // (48)
    const _Float16* __restrict__ Az,    // (48, 64) f16
    const _Float16* __restrict__ Av,    // (48, 64) f16
    float* __restrict__ out,            // (B, 48) row-major
    int Bsz, int ntiles) {
  const int lane  = threadIdx.x & 31;
  const int c     = lane & 15;   // = N column of B/C/D, = M row of A
  const int khalf = lane >> 4;   // selects K sub-block (A) / K row pair half (B)

  const int wave = blockIdx.x * (blockDim.x >> 5) + (threadIdx.x >> 5);
  const int nwav = gridDim.x * (blockDim.x >> 5);

  // ---- persistent A fragments (16x32 f16 A, contiguous k order, same as B):
  // lane m (0..15): K = khalf*16 + {0..15} in halves 0..15 (contiguous)
  v16h aZ[3][2], aV[3][2];
#pragma unroll
  for (int rt = 0; rt < 3; ++rt) {
#pragma unroll
    for (int ch = 0; ch < 2; ++ch) {
      const int row = rt * 16 + c;
      const int k1  = ch * 32 + khalf * 16;
      const int k2  = k1 + 8;
      const v8h zlo = *(const v8h*)(Az + row * KPAD + k1);
      const v8h zhi = *(const v8h*)(Az + row * KPAD + k2);
      const v8h vlo = *(const v8h*)(Av + row * KPAD + k1);
      const v8h vhi = *(const v8h*)(Av + row * KPAD + k2);
      v16h fz, fv;
#pragma unroll
      for (int i = 0; i < 8; ++i) {
        fz[i] = zlo[i]; fz[8 + i] = zhi[i];
        fv[i] = vlo[i]; fv[8 + i] = vhi[i];
      }
      aZ[rt][ch] = fz;
      aV[rt][ch] = fv;
    }
  }

  for (int t = wave; t < ntiles; t += nwav) {
    const int b0 = t * 16;
    const float* hrow = hidden + (size_t)(b0 + c) * Hdim;

    // ---- B fragment chunk 0: rows K = khalf*16 + 0..15 of sigmoid(hidden^T)
    // (k order used here, as in A: VGPR i, lane c holds K=2i,2i+1; lane c+16 holds K=16+2i,+1)
    v16h bf0;
    {
      const float* p  = hrow + khalf * 16;
      const f32x4 q0 = *(const f32x4*)(p);
      const f32x4 q1 = *(const f32x4*)(p + 4);
      const f32x4 q2 = *(const f32x4*)(p + 8);
      const f32x4 q3 = *(const f32x4*)(p + 12);
#pragma unroll
      for (int i = 0; i < 4; ++i) {
        bf0[i]      = (_Float16)sigf(q0[i]);
        bf0[4 + i]  = (_Float16)sigf(q1[i]);
        bf0[8 + i]  = (_Float16)sigf(q2[i]);
        bf0[12 + i] = (_Float16)sigf(q3[i]);
      }
    }
    // ---- B fragment chunk 1:
    //   khalf==0 lanes: K=32..47  -> sigmoid(hidden cols 32..47)
    //   khalf==1 lanes: K=48..55  -> x rows 0..7 ; K=56..63 -> 0
    v16h bf1;
    if (khalf == 0) {
      const float* p  = hrow + 32;
      const f32x4 q0 = *(const f32x4*)(p);
      const f32x4 q1 = *(const f32x4*)(p + 4);
      const f32x4 q2 = *(const f32x4*)(p + 8);
      const f32x4 q3 = *(const f32x4*)(p + 12);
#pragma unroll
      for (int i = 0; i < 4; ++i) {
        bf1[i]      = (_Float16)sigf(q0[i]);
        bf1[4 + i]  = (_Float16)sigf(q1[i]);
        bf1[8 + i]  = (_Float16)sigf(q2[i]);
        bf1[12 + i] = (_Float16)sigf(q3[i]);
      }
    } else {
#pragma unroll
      for (int k = 0; k < INdim; ++k)
        bf1[k] = (_Float16)x[(size_t)k * Bsz + (b0 + c)];
#pragma unroll
      for (int k = INdim; k < 16; ++k)
        bf1[k] = (_Float16)0.0f;
    }

#pragma unroll
    for (int rt = 0; rt < 3; ++rt) {
      const v8f zero = {};
      v8f accZ = __builtin_amdgcn_wmma_f32_16x16x32_f16(
          false, aZ[rt][0], false, bf0, (short)0, zero, false, false);
      accZ = __builtin_amdgcn_wmma_f32_16x16x32_f16(
          false, aZ[rt][1], false, bf1, (short)0, accZ, false, false);
      v8f accV = __builtin_amdgcn_wmma_f32_16x16x32_f16(
          false, aV[rt][0], false, bf0, (short)0, zero, false, false);
      accV = __builtin_amdgcn_wmma_f32_16x16x32_f16(
          false, aV[rt][1], false, bf1, (short)0, accV, false, false);

      // C/D layout: VGPR j -> M = j + 8*khalf, N = c. Rows contiguous per lane.
      const int m0 = rt * 16 + khalf * 8;
      const f32x4 vv0 = *(const f32x4*)(hrow + m0);
      const f32x4 vv1 = *(const f32x4*)(hrow + m0 + 4);
      const f32x4 bz0 = *(const f32x4*)(b_z + m0);
      const f32x4 bz1 = *(const f32x4*)(b_z + m0 + 4);
      const f32x4 bv0 = *(const f32x4*)(b_v + m0);
      const f32x4 bv1 = *(const f32x4*)(b_v + m0 + 4);
      f32x4 r0, r1;
#pragma unroll
      for (int j = 0; j < 4; ++j) {
        const float z0 = DTc * sigf(accZ[j] + bz0[j]);
        r0[j] = (1.0f - z0) * vv0[j] + DTc * (accV[j] + bv0[j]);
        const float z1 = DTc * sigf(accZ[4 + j] + bz1[j]);
        r1[j] = (1.0f - z1) * vv1[j] + DTc * (accV[4 + j] + bv1[j]);
      }
      float* orow = out + (size_t)(b0 + c) * Hdim + m0;
      *(f32x4*)(orow)     = r0;
      *(f32x4*)(orow + 4) = r1;
    }
  }
}

// ---------------------------------------------------------------------------
// Launcher. Input order (setup_inputs dict):
//   0 hidden (B,48) 1 x (8,B) 2 P (48,8) 3 b_v (48) 4 K (48,48) 5 C (48,48)
//   6 P_z (48,8) 7 b_z (48) 8 e_e (1) 9 e_i (1)
// ---------------------------------------------------------------------------
extern "C" void kernel_launch(void* const* d_in, const int* in_sizes, int n_in,
                              void* d_out, int out_size, void* d_ws, size_t ws_size,
                              hipStream_t stream) {
  const float* hidden = (const float*)d_in[0];
  const float* x      = (const float*)d_in[1];
  const float* P      = (const float*)d_in[2];
  const float* b_v    = (const float*)d_in[3];
  const float* Kw     = (const float*)d_in[4];
  const float* Cw     = (const float*)d_in[5];
  const float* P_z    = (const float*)d_in[6];
  const float* b_z    = (const float*)d_in[7];
  const float* e_e    = (const float*)d_in[8];
  const float* e_i    = (const float*)d_in[9];
  float* out = (float*)d_out;

  _Float16* Az = (_Float16*)d_ws;            // 48*64 f16 = 6 KB
  _Float16* Av = Az + Hdim * KPAD;           // next 6 KB

  const int Bsz    = in_sizes[0] / Hdim;     // 1048576
  const int ntiles = Bsz / 16;               // 65536 (B is a multiple of 16)

  dale_prep<<<1, 256, 0, stream>>>(Kw, Cw, P, P_z, e_e, e_i, Az, Av);

  const int threads = 256;                   // 8 waves / block
  const int blocks  = 4096;                  // grid-stride: 2 tiles per wave
  dale_main<<<blocks, threads, 0, stream>>>(hidden, x, b_v, b_z, Az, Av, out,
                                            Bsz, ntiles);
}
